// FLOT_46832323395799
// MI455X (gfx1250) — matrix operand order repaired and checked by reading, not compiled, not run on hardware
//
#include <hip/hip_runtime.h>
#include <hip/hip_bf16.h>

// ---------------------------------------------------------------------------
// FLOT (optimal-transport scene flow) for MI455X / gfx1250, wave32 + WMMA.
// B=2, N=4096, K=32.  All heavy GEMMs (set_conv MLPs, Sinkhorn cost matrix)
// run on v_wmma_f32_16x16x32_f16.  Grouped tensors and the NxN Kmat are never
// materialized in HBM: set_conv fuses gather+3 MLP layers+maxpool in LDS;
// Sinkhorn recomputes K[n,m] on the fly in two fused WMMA passes.
// Round-3: points packed as float4 (x,y,z,|p|^2) -> one aligned b128 load per
// point in the NxN loops (kills the v_dual_mov shuffle storm + norm recompute).
// ---------------------------------------------------------------------------

#define BB  2
#define NN  4096
#define KNB 32

typedef __attribute__((ext_vector_type(16))) _Float16 v16h;
typedef __attribute__((ext_vector_type(8)))  float    v8f;

__device__ __forceinline__ int lane_id() { return threadIdx.x & 31; }

// A-fragment (16xK, f16), row = lane&15.  ISA 7.12.2 16-bit A layout:
// lanes 0-15: K = {0..7, 16..23}, lanes 16-31: K = {8..15, 24..31} (+kbase).
__device__ __forceinline__ v16h ld_frag_a(const _Float16* base, int stride, int kbase) {
  int lane = lane_id();
  const uint32_t* row = (const uint32_t*)(base + (size_t)(lane & 15) * stride);
  int kb = (kbase + ((lane & 16) ? 8 : 0)) >> 1;   // in f16-pairs
  union { v16h v; uint32_t u[8]; } r;
#pragma unroll
  for (int i = 0; i < 4; ++i) { r.u[i] = row[kb + i]; r.u[4 + i] = row[kb + 8 + i]; }
  return r.v;
}

// B-fragment (Kx16, f16) from column-major storage [col][k], col = lane&15.
// lanes 0-15: K = {0..7, 16..23}, lanes 16-31: K = {8..15, 24..31} (+kbase).
__device__ __forceinline__ v16h ld_frag_b(const _Float16* base, int stride, int kbase) {
  int lane = lane_id();
  const uint32_t* col = (const uint32_t*)(base + (size_t)(lane & 15) * stride);
  int kb = (kbase + ((lane & 16) ? 8 : 0)) >> 1;
  union { v16h v; uint32_t u[8]; } r;
#pragma unroll
  for (int i = 0; i < 4; ++i) { r.u[i] = col[kb + i]; r.u[4 + i] = col[kb + 8 + i]; }
  return r.v;
}

// ---------------------------------------------------------------------------
// Pack points as (x, y, z, |p|^2): one aligned b128 load per point afterwards.
// ---------------------------------------------------------------------------
__global__ __launch_bounds__(256) void pack_xyz_kernel(const float* __restrict__ pc,
                                                       float4* __restrict__ pcq) {
  int b = blockIdx.y;
  int i = blockIdx.x * 256 + threadIdx.x;
  float x = pc[((size_t)b * NN + i) * 3 + 0];
  float y = pc[((size_t)b * NN + i) * 3 + 1];
  float z = pc[((size_t)b * NN + i) * 3 + 2];
  pcq[(size_t)b * NN + i] = make_float4(x, y, z, x * x + y * y + z * z);
}

// ---------------------------------------------------------------------------
// kNN: one thread per query, candidate tiles staged in LDS, top-32 insertion.
// ---------------------------------------------------------------------------
__global__ __launch_bounds__(256) void knn_kernel(const float4* __restrict__ pcq,
                                                  int* __restrict__ idx) {
  int b = blockIdx.y;
  int q = blockIdx.x * 256 + threadIdx.x;
  float4 qp = pcq[(size_t)b * NN + q];
  float bd[KNB]; int bi[KNB];
#pragma unroll
  for (int i = 0; i < KNB; ++i) { bd[i] = 3.0e38f; bi[i] = 0; }
  __shared__ float4 sp[256];
  for (int jb = 0; jb < NN; jb += 256) {
    __syncthreads();
    sp[threadIdx.x] = pcq[(size_t)b * NN + jb + threadIdx.x];
    __syncthreads();
    for (int jj = 0; jj < 256; ++jj) {
      float4 cp = sp[jj];
      float dx = qp.x - cp.x, dy = qp.y - cp.y, dz = qp.z - cp.z;
      float d = dx * dx + dy * dy + dz * dz;
      if (d < bd[KNB - 1]) {
        int p = KNB - 1;
        while (p > 0 && bd[p - 1] > d) { bd[p] = bd[p - 1]; bi[p] = bi[p - 1]; --p; }
        bd[p] = d; bi[p] = jb + jj;
      }
    }
  }
#pragma unroll
  for (int k = 0; k < KNB; ++k) idx[((size_t)b * NN + q) * KNB + k] = bi[k];
}

// ---------------------------------------------------------------------------
// One MLP layer over 128 LDS columns with WMMA 16x16x32 f16->f32.
// src/dst: [128][STR] col-major f16, sW: [CH][STR] row-major f16.
// Epilogue packs f16 pairs -> single b32 LDS stores.
// ---------------------------------------------------------------------------
__device__ __forceinline__ void mlp_layer(const _Float16* src, _Float16* dst,
                                          const _Float16* sWp, const float* sBias,
                                          int CHv, int STRv, int ksteps, int tid) {
  const int CT = 8;                 // 128 cols / 16
  int RT   = CHv >> 4;
  int wave = tid >> 5;
  int lane = tid & 31;
  int hi   = (lane & 16) ? 8 : 0;
  for (int t = wave; t < CT * RT; t += 8) {
    int ct = t & 7, rt = t >> 3;
    v8f acc = {0.f, 0.f, 0.f, 0.f, 0.f, 0.f, 0.f, 0.f};
    for (int ks = 0; ks < ksteps; ++ks) {
      v16h a  = ld_frag_a(sWp + rt * 16 * STRv, STRv, ks * 32);
      v16h bm = ld_frag_b(src + ct * 16 * STRv, STRv, ks * 32);
      acc = __builtin_amdgcn_wmma_f32_16x16x32_f16(false, a, false, bm,
                                                   (short)0, acc, false, false);
    }
    int col = ct * 16 + (lane & 15);
    int m0  = rt * 16 + hi;                       // even; 8 consecutive channels
    _Float16* dcol = dst + col * STRv + m0;
    union { v8f v; float f[8]; } u; u.v = acc;
#pragma unroll
    for (int w = 0; w < 4; ++w) {
      float x0 = u.f[2 * w]     + sBias[m0 + 2 * w];
      float x1 = u.f[2 * w + 1] + sBias[m0 + 2 * w + 1];
      x0 = x0 > 0.f ? x0 : 0.f;                   // ReLU
      x1 = x1 > 0.f ? x1 : 0.f;
      union { _Float16 h[2]; uint32_t u32; } pk;
      pk.h[0] = (_Float16)x0; pk.h[1] = (_Float16)x1;
      *(uint32_t*)(dcol + 2 * w) = pk.u32;        // 4B-aligned packed store
    }
  }
}

__device__ __forceinline__ void load_weights(const float* Wg, const float* Bg,
                                             int cinL, int CHv, int STRv,
                                             _Float16* sWp, float* sBp, int tid) {
  for (int t = tid; t < CHv * CHv; t += 256) {
    int r = t / CHv, c = t % CHv;
    sWp[r * STRv + c] = (_Float16)((c < cinL) ? Wg[r * cinL + c] : 0.f);
  }
  for (int t = tid; t < CHv; t += 256) sBp[t] = Bg[t];
}

// ---------------------------------------------------------------------------
// Fused set_conv: gather(rel-xyz || feat) -> 3x (1x1 conv + ReLU) -> maxpool.
// 4 points (128 neighbor-columns) per workgroup; all intermediates in LDS.
// ---------------------------------------------------------------------------
template <int CIN, int CH>
__global__ __launch_bounds__(256) void setconv_kernel(
    const float* __restrict__ pc,       // [B,N,3]
    const float* __restrict__ featin,   // [B,N,3] if feat_bn3 else [B,CIN,N]
    int feat_bn3,
    const int* __restrict__ idxg,       // [B,N,KNB]
    const float* __restrict__ W1, const float* __restrict__ B1,
    const float* __restrict__ W2, const float* __restrict__ B2,
    const float* __restrict__ W3, const float* __restrict__ B3,
    float* __restrict__ out)            // [B,CH,N]
{
  constexpr int CIN_T  = CIN + 3;
  constexpr int KP0    = ((CIN_T + 31) / 32) * 32;
  static_assert(KP0 <= CH, "k-padding must fit in channel buffer");
  constexpr int STR    = CH + 8;              // half-padding: kills LDS bank conflicts
  constexpr int TILE_P = 4;
  constexpr int COLS   = TILE_P * KNB;        // 128

  __shared__ _Float16 sW[CH * STR];
  __shared__ float    sBias[CH];
  __shared__ _Float16 sF0[COLS * STR];
  __shared__ _Float16 sF1[COLS * STR];
  __shared__ int      sIdx[COLS];

  int b   = blockIdx.y;
  int n0  = blockIdx.x * TILE_P;
  int tid = threadIdx.x;

  for (int t = tid; t < COLS; t += 256)
    sIdx[t] = idxg[((size_t)b * NN + n0 + t / KNB) * KNB + (t % KNB)];
  load_weights(W1, B1, CIN_T, CH, STR, sW, sBias, tid);
  __syncthreads();

  // gather: channels [0..2]=rel xyz, [3..CIN_T)=grouped feats, rest zero-pad
  for (int t = tid; t < COLS * KP0; t += 256) {
    int col = t / KP0, c = t % KP0;
    int p = col / KNB;
    int j = sIdx[col];
    int n = n0 + p;
    float v = 0.f;
    if (c < 3) {
      v = pc[((size_t)b * NN + j) * 3 + c] - pc[((size_t)b * NN + n) * 3 + c];
    } else if (c < CIN_T) {
      int cc = c - 3;
      v = feat_bn3 ? featin[((size_t)b * NN + j) * 3 + cc]
                   : featin[((size_t)b * CIN + cc) * NN + j];
    }
    sF0[col * STR + c] = (_Float16)v;
  }
  __syncthreads();

  mlp_layer(sF0, sF1, sW, sBias, CH, STR, KP0 / 32, tid);
  __syncthreads();
  load_weights(W2, B2, CH, CH, STR, sW, sBias, tid);
  __syncthreads();
  mlp_layer(sF1, sF0, sW, sBias, CH, STR, CH / 32, tid);
  __syncthreads();
  load_weights(W3, B3, CH, CH, STR, sW, sBias, tid);
  __syncthreads();
  mlp_layer(sF0, sF1, sW, sBias, CH, STR, CH / 32, tid);
  __syncthreads();

  // maxpool over KNB neighbors
  for (int t = tid; t < TILE_P * CH; t += 256) {
    int p = t / CH, c = t % CH;
    float m = -3.0e38f;
#pragma unroll 4
    for (int kk = 0; kk < KNB; ++kk) {
      float v = (float)sF1[(p * KNB + kk) * STR + c];
      m = v > m ? v : m;
    }
    out[((size_t)b * CH + c) * NN + n0 + p] = m;
  }
}

// ---------------------------------------------------------------------------
// L2-normalize 128-ch features, transpose to [B,N,128] f16 for WMMA fragments.
// ---------------------------------------------------------------------------
__global__ __launch_bounds__(256) void norm_kernel(const float* __restrict__ f,
                                                   _Float16* __restrict__ fh) {
  int b = blockIdx.y;
  int i = blockIdx.x * 256 + threadIdx.x;
  float s = 0.f;
  for (int c = 0; c < 128; ++c) {
    float v = f[((size_t)b * 128 + c) * NN + i];
    s += v * v;
  }
  float r = rsqrtf(s + 1e-8f);
  for (int c = 0; c < 128; ++c) {
    float v = f[((size_t)b * 128 + c) * NN + i];
    fh[((size_t)b * NN + i) * 128 + c] = (_Float16)(v * r);
  }
}

// ---------------------------------------------------------------------------
// Sinkhorn pass 1: colsum[m] = sum_n K[n,m] (a0 uniform) -> b[m].
// K recomputed on the fly: 16x16 WMMA dot tile + dist gate + exp.
// ---------------------------------------------------------------------------
__global__ __launch_bounds__(256) void sink_colsum_kernel(
    const _Float16* __restrict__ f1h, const _Float16* __restrict__ f2h,
    const float4* __restrict__ pcq1, const float4* __restrict__ pcq2,
    const float* __restrict__ gamma, const float* __restrict__ epsilon,
    float* __restrict__ bvec)
{
  __shared__ float colacc[128];
  int b   = blockIdx.y;
  int m0  = blockIdx.x * 128;
  int tid = threadIdx.x;
  if (tid < 128) colacc[tid] = 0.f;
  __syncthreads();

  int wave = tid >> 5, lane = tid & 31;
  int mT = m0 + wave * 16;
  int lm = wave * 16 + (lane & 15);
  int m  = m0 + lm;
  int hi = (lane & 16) ? 8 : 0;
  float eps = __expf(epsilon[0]) + 0.03f;
  float neg_inv_eps = -1.0f / eps;        // hoisted: no per-element division

  v16h bm[4];
#pragma unroll
  for (int ks = 0; ks < 4; ++ks)
    bm[ks] = ld_frag_b(f2h + ((size_t)b * NN + mT) * 128, 128, ks * 32);

  float4 q2 = pcq2[(size_t)b * NN + m];   // (x,y,z,|p|^2): single b128 load

  for (int nb = 0; nb < NN; nb += 16) {
    if (nb + 16 < NN)  // keep the A stream hot in L2/L0 (global_prefetch_b8)
      __builtin_prefetch(f1h + ((size_t)b * NN + nb + 16) * 128, 0, 3);
    v8f acc = {0.f, 0.f, 0.f, 0.f, 0.f, 0.f, 0.f, 0.f};
#pragma unroll
    for (int ks = 0; ks < 4; ++ks) {
      v16h a = ld_frag_a(f1h + ((size_t)b * NN + nb) * 128, 128, ks * 32);
      acc = __builtin_amdgcn_wmma_f32_16x16x32_f16(false, a, false, bm[ks],
                                                   (short)0, acc, false, false);
    }
    union { v8f v; float f[8]; } u; u.v = acc;
    float csum = 0.f;
#pragma unroll
    for (int v = 0; v < 8; ++v) {
      int n = nb + v + hi;
      float4 q1 = pcq1[(size_t)b * NN + n];
      float dsq = q1.w + q2.w - 2.f * (q1.x * q2.x + q1.y * q2.y + q1.z * q2.z);
      float e  = __expf((1.f - u.f[v]) * neg_inv_eps);
      float Kv = (dsq < 100.f) ? e : 0.f;          // branchless -> v_cndmask
      csum += Kv;
    }
    atomicAdd(&colacc[lm], csum);
  }
  __syncthreads();
  if (tid < 128) {
    float gam   = __expf(gamma[0]);
    float power = gam / (gam + eps);
    float KTa   = colacc[tid] * (1.f / NN);           // a0 = 1/n1
    bvec[(size_t)b * NN + m0 + tid] = powf((1.f / NN) / (KTa + 1e-8f), power);
  }
}

// ---------------------------------------------------------------------------
// Sinkhorn pass 2: Kb[n] = sum_m K[n,m] b[m], S[n] = sum_m K[n,m] b[m] pc2[m];
// then a, row_sum, flow = pc1 - (a*S)/(a*Kb + 1e-8).  flow stored [B,3,N].
// ---------------------------------------------------------------------------
__global__ __launch_bounds__(256) void sink_flow_kernel(
    const _Float16* __restrict__ f1h, const _Float16* __restrict__ f2h,
    const float4* __restrict__ pcq1, const float4* __restrict__ pcq2,
    const float* __restrict__ gamma, const float* __restrict__ epsilon,
    const float* __restrict__ bvec, float* __restrict__ flow)
{
  __shared__ float sKb[128];
  __shared__ float sS[128 * 3];
  int b   = blockIdx.y;
  int n0  = blockIdx.x * 128;
  int tid = threadIdx.x;
  if (tid < 128) {
    sKb[tid] = 0.f;
    sS[tid * 3 + 0] = 0.f; sS[tid * 3 + 1] = 0.f; sS[tid * 3 + 2] = 0.f;
  }
  __syncthreads();

  int wave = tid >> 5, lane = tid & 31;
  int nT = n0 + wave * 16;
  int hi = (lane & 16) ? 8 : 0;
  float eps = __expf(epsilon[0]) + 0.03f;
  float neg_inv_eps = -1.0f / eps;

  v16h am[4];
#pragma unroll
  for (int ks = 0; ks < 4; ++ks)
    am[ks] = ld_frag_a(f1h + ((size_t)b * NN + nT) * 128, 128, ks * 32);

  float4 q1v[8];
#pragma unroll
  for (int v = 0; v < 8; ++v)
    q1v[v] = pcq1[(size_t)b * NN + nT + v + hi];

  float aKb[8], aSx[8], aSy[8], aSz[8];
#pragma unroll
  for (int v = 0; v < 8; ++v) { aKb[v] = 0.f; aSx[v] = 0.f; aSy[v] = 0.f; aSz[v] = 0.f; }

  for (int mb = 0; mb < NN; mb += 16) {
    if (mb + 16 < NN)
      __builtin_prefetch(f2h + ((size_t)b * NN + mb + 16) * 128, 0, 3);
    v8f acc = {0.f, 0.f, 0.f, 0.f, 0.f, 0.f, 0.f, 0.f};
#pragma unroll
    for (int ks = 0; ks < 4; ++ks) {
      v16h bm = ld_frag_b(f2h + ((size_t)b * NN + mb) * 128, 128, ks * 32);
      acc = __builtin_amdgcn_wmma_f32_16x16x32_f16(false, am[ks], false, bm,
                                                   (short)0, acc, false, false);
    }
    int m = mb + (lane & 15);
    float bv  = bvec[(size_t)b * NN + m];
    float4 q2 = pcq2[(size_t)b * NN + m];
    union { v8f v; float f[8]; } u; u.v = acc;
#pragma unroll
    for (int v = 0; v < 8; ++v) {
      float dsq = q1v[v].w + q2.w
                - 2.f * (q1v[v].x * q2.x + q1v[v].y * q2.y + q1v[v].z * q2.z);
      float e  = __expf((1.f - u.f[v]) * neg_inv_eps);
      float Kv = (dsq < 100.f) ? e : 0.f;
      float w = Kv * bv;
      aKb[v] += w; aSx[v] += w * q2.x; aSy[v] += w * q2.y; aSz[v] += w * q2.z;
    }
  }
#pragma unroll
  for (int v = 0; v < 8; ++v) {
    int li = wave * 16 + v + hi;
    atomicAdd(&sKb[li], aKb[v]);
    atomicAdd(&sS[li * 3 + 0], aSx[v]);
    atomicAdd(&sS[li * 3 + 1], aSy[v]);
    atomicAdd(&sS[li * 3 + 2], aSz[v]);
  }
  __syncthreads();
  if (tid < 128) {
    int n = n0 + tid;
    float gam   = __expf(gamma[0]);
    float power = gam / (gam + eps);
    float Kb  = sKb[tid];
    float a   = powf((1.f / NN) / (Kb + 1e-8f), power);
    float row = a * Kb;
    float inv = 1.f / (row + 1e-8f);
    float4 q1 = pcq1[(size_t)b * NN + n];
    float p[3] = {q1.x, q1.y, q1.z};
#pragma unroll
    for (int d = 0; d < 3; ++d) {
      float pin = a * sS[tid * 3 + d] * inv;
      flow[((size_t)b * 3 + d) * NN + n] = p[d] - pin;
    }
  }
}

// ---------------------------------------------------------------------------
// Final: sf = ot_flow + (fc_w @ g + fc_b), output [B,N,3].
// ---------------------------------------------------------------------------
__global__ __launch_bounds__(256) void final_kernel(
    const float* __restrict__ flow, const float* __restrict__ g,
    const float* __restrict__ fcw, const float* __restrict__ fcb,
    float* __restrict__ outp)
{
  int b = blockIdx.y;
  int i = blockIdx.x * 256 + threadIdx.x;
  float r0 = fcb[0], r1 = fcb[1], r2 = fcb[2];
  for (int c = 0; c < 128; ++c) {
    float gv = g[((size_t)b * 128 + c) * NN + i];
    r0 += fcw[0 * 128 + c] * gv;
    r1 += fcw[1 * 128 + c] * gv;
    r2 += fcw[2 * 128 + c] * gv;
  }
  outp[((size_t)b * NN + i) * 3 + 0] = flow[((size_t)b * 3 + 0) * NN + i] + r0;
  outp[((size_t)b * NN + i) * 3 + 1] = flow[((size_t)b * 3 + 1) * NN + i] + r1;
  outp[((size_t)b * NN + i) * 3 + 2] = flow[((size_t)b * 3 + 2) * NN + i] + r2;
}

// ---------------------------------------------------------------------------
// Host orchestration.  Input leaf order (setup_inputs insertion order):
//   0:pc1 1:pc2 | 2..7:sa1(W,b)x3 | 8..13:sa2 | 14..19:sa3 |
//   20..25:rf1 | 26..31:rf2 | 32..37:rf3 | 38:fc_w 39:fc_b 40:gamma 41:epsilon
// ---------------------------------------------------------------------------
extern "C" void kernel_launch(void* const* d_in, const int* in_sizes, int n_in,
                              void* d_out, int out_size, void* d_ws, size_t ws_size,
                              hipStream_t stream) {
  (void)in_sizes; (void)n_in; (void)out_size; (void)ws_size;
  auto F = [&](int i) { return (const float*)d_in[i]; };
  const float* pc1 = F(0);
  const float* pc2 = F(1);

  // workspace carve (~14.4 MB total)
  char* w = (char*)d_ws;
  float* fA = (float*)w;             w += sizeof(float) * (size_t)BB * 128 * NN;   // 4 MB
  float* fB = (float*)w;             w += sizeof(float) * (size_t)BB * 128 * NN;   // 4 MB
  _Float16* f1h = (_Float16*)w;      w += sizeof(_Float16) * (size_t)BB * NN * 128;// 2 MB
  _Float16* f2h = (_Float16*)w;      w += sizeof(_Float16) * (size_t)BB * NN * 128;// 2 MB
  float* bvec = (float*)w;           w += sizeof(float) * (size_t)BB * NN;
  float* flow = (float*)w;           w += sizeof(float) * (size_t)BB * 3 * NN;
  int* idx1 = (int*)w;               w += sizeof(int) * (size_t)BB * NN * KNB;     // 1 MB
  int* idx2 = (int*)w;               w += sizeof(int) * (size_t)BB * NN * KNB;     // 1 MB
  float4* pcq1 = (float4*)w;         w += sizeof(float4) * (size_t)BB * NN;        // 128 KB
  float4* pcq2 = (float4*)w;         w += sizeof(float4) * (size_t)BB * NN;        // 128 KB

  dim3 blk(256);
  dim3 gN(NN / 256, BB);
  dim3 gSC(NN / 4, BB);
  dim3 gSK(NN / 128, BB);

  pack_xyz_kernel<<<gN, blk, 0, stream>>>(pc1, pcq1);
  pack_xyz_kernel<<<gN, blk, 0, stream>>>(pc2, pcq2);
  knn_kernel<<<gN, blk, 0, stream>>>(pcq1, idx1);
  knn_kernel<<<gN, blk, 0, stream>>>(pcq2, idx2);

  // feature pyramid, cloud 1
  setconv_kernel<3, 32>  <<<gSC, blk, 0, stream>>>(pc1, pc1, 1, idx1, F(2), F(3), F(4), F(5), F(6), F(7), fA);
  setconv_kernel<32, 64> <<<gSC, blk, 0, stream>>>(pc1, fA, 0, idx1, F(8), F(9), F(10), F(11), F(12), F(13), fB);
  setconv_kernel<64, 128><<<gSC, blk, 0, stream>>>(pc1, fB, 0, idx1, F(14), F(15), F(16), F(17), F(18), F(19), fA);
  norm_kernel<<<gN, blk, 0, stream>>>(fA, f1h);

  // feature pyramid, cloud 2 (reuse fA/fB)
  setconv_kernel<3, 32>  <<<gSC, blk, 0, stream>>>(pc2, pc2, 1, idx2, F(2), F(3), F(4), F(5), F(6), F(7), fA);
  setconv_kernel<32, 64> <<<gSC, blk, 0, stream>>>(pc2, fA, 0, idx2, F(8), F(9), F(10), F(11), F(12), F(13), fB);
  setconv_kernel<64, 128><<<gSC, blk, 0, stream>>>(pc2, fB, 0, idx2, F(14), F(15), F(16), F(17), F(18), F(19), fA);
  norm_kernel<<<gN, blk, 0, stream>>>(fA, f2h);

  // fused Sinkhorn (max_iter = 1) without materializing Kmat
  sink_colsum_kernel<<<gSK, blk, 0, stream>>>(f1h, f2h, pcq1, pcq2, F(40), F(41), bvec);
  sink_flow_kernel  <<<gSK, blk, 0, stream>>>(f1h, f2h, pcq1, pcq2, F(40), F(41), bvec, flow);

  // refinement pyramid on flow (layout [B,3,N] -> feat_bn3 = 0)
  setconv_kernel<3, 32>  <<<gSC, blk, 0, stream>>>(pc1, flow, 0, idx1, F(20), F(21), F(22), F(23), F(24), F(25), fA);
  setconv_kernel<32, 64> <<<gSC, blk, 0, stream>>>(pc1, fA, 0, idx1, F(26), F(27), F(28), F(29), F(30), F(31), fB);
  setconv_kernel<64, 128><<<gSC, blk, 0, stream>>>(pc1, fB, 0, idx1, F(32), F(33), F(34), F(35), F(36), F(37), fA);

  final_kernel<<<gN, blk, 0, stream>>>(flow, fA, F(38), F(39), (float*)d_out);
}
